// NeuralTPP_90056874262803
// MI455X (gfx1250) — matrix operand order repaired and checked by reading, not compiled, not run on hardware
//
#include <hip/hip_runtime.h>
#include <hip/hip_bf16.h>

typedef __attribute__((ext_vector_type(16))) _Float16 v16h;
typedef __attribute__((ext_vector_type(8)))  float    v8f;

#define Bsz 256
#define Ssz 2048
#define Hsz 128
#define ROWQ 17    // padded row stride in float4 units (17*16 B = 272 B): bank-conflict-free

union Frag16 { v16h v; float4 q[2]; _Float16 h[16]; };
union Pack8  { float4 q; _Float16 h[8]; };

template<int N> struct IC { static constexpr int v = N; };

// ---- native CDNA5 transcendentals (probe-confirmed: v_tanh_f32 / v_exp_f32 / v_log_f32) ----
#if __has_builtin(__builtin_amdgcn_tanhf)
__device__ __forceinline__ float fast_tanh(float x) { return __builtin_amdgcn_tanhf(x); }
#elif __has_builtin(__builtin_amdgcn_tanh_f32)
__device__ __forceinline__ float fast_tanh(float x) { return __builtin_amdgcn_tanh_f32(x); }
#else
__device__ __forceinline__ float fast_tanh(float x) { return tanhf(x); }
#endif

#if __has_builtin(__builtin_amdgcn_logf)
__device__ __forceinline__ float fast_ln(float x) { return __builtin_amdgcn_logf(x) * 0.69314718055994531f; }
#else
__device__ __forceinline__ float fast_ln(float x) { return log2f(x) * 0.69314718055994531f; }
#endif

#if __has_builtin(__builtin_amdgcn_exp2f)
__device__ __forceinline__ float fast_exp(float x) { return __builtin_amdgcn_exp2f(x * 1.4426950408889634f); }
#else
__device__ __forceinline__ float fast_exp(float x) { return exp2f(x * 1.4426950408889634f); }
#endif

// One workgroup = 16 batch rows, 8 waves; wave w owns hidden columns [16w, 16w+16).
// Transposed product per step: D'(n,m) = sum_k W_hh[n][k] * h[m][k]
//   A = W_hh tile (constant, register A-fragments) ; B = h^T (native float4 LDS reads)
// Double-buffered state with compile-time parity; one barrier per step.
__launch_bounds__(256, 1)
__global__ void tpp_scan_kernel(const float* __restrict__ times,
                                const int*   __restrict__ mask,
                                const float* __restrict__ W_ih,
                                const float* __restrict__ W_hh,
                                const float* __restrict__ b_ih,
                                const float* __restrict__ b_hh,
                                const float* __restrict__ W_lin,
                                const float* __restrict__ b_lin,
                                float*       __restrict__ out)
{
    __shared__ float4   hbuf4[2][16 * ROWQ]; // h state: 8 f16 per float4, padded rows
    __shared__ float2   tstage[2][16];       // (t, log t) per batch row
    __shared__ float2   hacc[2][16];         // per-row (hist.Wl0, hist.Wl1) accumulators
    __shared__ int      ntrue_s[16];         // per-row True count of prefix mask

    const int tid    = threadIdx.x;
    const int wave   = tid >> 5;           // 0..7 : N-tile of hidden columns
    const int lane   = tid & 31;
    const int lanelo = lane & 15;          // = batch row m of this lane's D' column
    const int hi     = lane >> 4;
    const int hi8    = hi * 8;
    const int bm     = blockIdx.x * 16;    // batch-row base
    const int ng     = wave * 16 + lanelo; // W_hh row (hidden col) for A-fragments
    const int osel   = wave & 1;           // head output this wave reduces (0=mu,1=lsg)
    const int hch    = wave >> 1;          // head K-chunk this wave reduces
    const int rowb   = lanelo * ROWQ;      // this lane's h row base (float4 units)

    // ---- init LDS ----
    for (int i = tid; i < 16 * ROWQ; i += 256) hbuf4[0][i] = make_float4(0.f, 0.f, 0.f, 0.f);
    if (tid < 16) { hacc[0][tid] = make_float2(0.0f, 0.0f); hacc[1][tid] = make_float2(0.0f, 0.0f); }

    // ---- n_true per row (int4-vectorized): wave w counts rows 2w / 2w+1 ----
    {
        const int row = bm + 2 * wave + hi;
        const int4* mp = reinterpret_cast<const int4*>(mask + row * Ssz);
        int pc = 0;
        for (int i = lanelo; i < Ssz / 4; i += 16) {
            const int4 v = mp[i];
            pc += (v.x != 0) + (v.y != 0) + (v.z != 0) + (v.w != 0);
        }
        pc += __shfl_xor(pc, 8, 32);
        pc += __shfl_xor(pc, 4, 32);
        pc += __shfl_xor(pc, 2, 32);
        pc += __shfl_xor(pc, 1, 32);
        if (lanelo == 0) ntrue_s[2 * wave + hi] = pc;
    }

    // ---- stage (t, log t) for step 0 ----
    if (wave == 7 && lane < 16) {
        const float t0 = fmaxf(times[(bm + lanelo) * Ssz + 0], 1e-8f);
        tstage[0][lanelo] = make_float2(t0, fast_ln(t0));
    }

    const float blin0 = b_lin[0];
    const float blin1 = b_lin[1];

    // ---- per-lane x_proj constants for the 8 hidden cols n = 16w + hi8 + r ----
    float wi0r[8], wi1r[8], bsr[8];
    {
        const float4* ip = reinterpret_cast<const float4*>(&W_ih[(16 * wave + hi8) * 2]);
        #pragma unroll
        for (int i = 0; i < 4; ++i) {
            const float4 q = ip[i];
            wi0r[2 * i]     = q.x;  wi1r[2 * i]     = q.y;
            wi0r[2 * i + 1] = q.z;  wi1r[2 * i + 1] = q.w;
        }
        const float4* b1 = reinterpret_cast<const float4*>(&b_ih[16 * wave + hi8]);
        const float4* b2 = reinterpret_cast<const float4*>(&b_hh[16 * wave + hi8]);
        #pragma unroll
        for (int i = 0; i < 2; ++i) {
            const float4 u = b1[i], v = b2[i];
            bsr[4 * i + 0] = u.x + v.x;  bsr[4 * i + 1] = u.y + v.y;
            bsr[4 * i + 2] = u.z + v.z;  bsr[4 * i + 3] = u.w + v.w;
        }
    }

    // ---- A fragments: W_hh tile, constant in VGPRs ----
    // A half j of chunk c holds k = 32c + 16*hi + j, row M = ng.
    Frag16 afr[4];
    #pragma unroll
    for (int c = 0; c < 4; ++c) {
        const float4* wp = reinterpret_cast<const float4*>(&W_hh[ng * Hsz + 32 * c + 16 * hi]);
        const float4 u0 = wp[0], u1 = wp[1], u2 = wp[2], u3 = wp[3];
        afr[c].h[0]  = (_Float16)u0.x; afr[c].h[1]  = (_Float16)u0.y;
        afr[c].h[2]  = (_Float16)u0.z; afr[c].h[3]  = (_Float16)u0.w;
        afr[c].h[4]  = (_Float16)u1.x; afr[c].h[5]  = (_Float16)u1.y;
        afr[c].h[6]  = (_Float16)u1.z; afr[c].h[7]  = (_Float16)u1.w;
        afr[c].h[8]  = (_Float16)u2.x; afr[c].h[9]  = (_Float16)u2.y;
        afr[c].h[10] = (_Float16)u2.z; afr[c].h[11] = (_Float16)u2.w;
        afr[c].h[12] = (_Float16)u3.x; afr[c].h[13] = (_Float16)u3.y;
        afr[c].h[14] = (_Float16)u3.z; afr[c].h[15] = (_Float16)u3.w;
    }

    // ---- W_lin slice matching this wave's head chunk (B half j <-> k = 32*hch+16*hi+j) ----
    float wlq[16];
    {
        const float4* lp = reinterpret_cast<const float4*>(&W_lin[osel * Hsz + 32 * hch + 16 * hi]);
        #pragma unroll
        for (int i = 0; i < 4; ++i) {
            const float4 q = lp[i];
            wlq[4 * i + 0] = q.x; wlq[4 * i + 1] = q.y;
            wlq[4 * i + 2] = q.z; wlq[4 * i + 3] = q.w;
        }
    }

    float acc     = 0.0f;   // per-row log-likelihood (wave 0, lanes 0..15)
    float lt_prev = 0.0f;   // log t of previous step (wave 0 stats pipeline)

    __syncthreads();

    // ---- one scan step; P = read-buffer parity (compile-time) ----
    auto step = [&](int s, auto pc) {
        constexpr int P = decltype(pc)::v;
        constexpr int Q = P ^ 1;

        const float2 tl = tstage[P][lanelo];       // this lane's (t, log t)

        // B fragments = h^T from read buffer: direct float4 LDS indexing -> ds_load_b128
        Frag16 b[4];
        #pragma unroll
        for (int c = 0; c < 4; ++c) {
            b[c].q[0] = hbuf4[P][rowb + 4 * c + 2 * hi];
            b[c].q[1] = hbuf4[P][rowb + 4 * c + 2 * hi + 1];
        }

        // C operand = x_proj(n = 16w+hi8+r, m = lanelo), two accumulation chains
        v8f d0, d1 = {};
        #pragma unroll
        for (int r = 0; r < 8; ++r)
            d0[r] = fmaf(tl.x, wi0r[r], fmaf(tl.y, wi1r[r], bsr[r]));

        // Head partials for position s -> hacc[P] (consumed next step)
        {
            float p0 = 0.0f, p1 = 0.0f;
            #pragma unroll
            for (int j = 0; j < 16; j += 2) {
                p0 = fmaf((float)b[hch].h[j],     wlq[j],     p0);
                p1 = fmaf((float)b[hch].h[j + 1], wlq[j + 1], p1);
            }
            float pr = p0 + p1;
            pr += __shfl_xor(pr, 16, 32);          // lanes m, m+16 jointly hold row m
            if (lane < 16)
                atomicAdd(osel ? &hacc[P][lanelo].y : &hacc[P][lanelo].x, pr);
        }

        // h_s = tanh(x_proj + h_{s-1} W_hh^T) -> write buffer
        d0 = __builtin_amdgcn_wmma_f32_16x16x32_f16(false, afr[0].v, false, b[0].v, (short)0, d0, false, false);
        d1 = __builtin_amdgcn_wmma_f32_16x16x32_f16(false, afr[1].v, false, b[1].v, (short)0, d1, false, false);
        d0 = __builtin_amdgcn_wmma_f32_16x16x32_f16(false, afr[2].v, false, b[2].v, (short)0, d0, false, false);
        d1 = __builtin_amdgcn_wmma_f32_16x16x32_f16(false, afr[3].v, false, b[3].v, (short)0, d1, false, false);

        Pack8 st;
        #pragma unroll
        for (int r = 0; r < 8; ++r)
            st.h[r] = (_Float16)fast_tanh(d0[r] + d1[r]);
        hbuf4[Q][rowb + 2 * wave + hi] = st.q;      // single ds_store_b128

        // Wave 0: stats for position s-1 from hacc[Q] (sealed by last barrier); reset it
        if (wave == 0 && lane < 16) {
            const float2 hm = hacc[Q][lane];
            hacc[Q][lane] = make_float2(0.0f, 0.0f);
            if (s > 0) {
                const float mu  = hm.x + blin0;
                const float lsg = hm.y + blin1;                      // log sigma
                const float z   = (lt_prev - mu) * fast_exp(-lsg);   // (log t - mu)/sigma
                const int   nt1 = ntrue_s[lane] - 1;                 // last-True index
                const int   pos = s - 1;
                if (pos < nt1)
                    acc += -lt_prev - lsg - 0.91893853320467274f - 0.5f * z * z;
                if (pos == nt1 && (nt1 + 1) < Ssz)                   // log survival
                    acc += logf(0.5f * erfcf(z * 0.70710678118654752f));
            }
            lt_prev = tl.y;
        }

        // Wave 7: stage (t, log t) for step s+1 into the other buffer
        if (wave == 7 && lane < 16 && (s + 1) < Ssz) {
            const float tn = fmaxf(times[(bm + lanelo) * Ssz + s + 1], 1e-8f);
            tstage[Q][lanelo] = make_float2(tn, fast_ln(tn));
        }

        __syncthreads();   // single barrier per step seals all cross-step deps
    };

    for (int s = 0; s < Ssz; s += 2) {
        step(s,     IC<0>{});
        step(s + 1, IC<1>{});
    }

    // Drain: position S-1 is never needed (n_true-1 <= S-2; all-True case contributes 0)
    if (wave == 0 && lane < 16) out[bm + lane] = acc;
}

extern "C" void kernel_launch(void* const* d_in, const int* in_sizes, int n_in,
                              void* d_out, int out_size, void* d_ws, size_t ws_size,
                              hipStream_t stream) {
    (void)in_sizes; (void)n_in; (void)out_size; (void)d_ws; (void)ws_size;
    const float* times = (const float*)d_in[0];
    const int*   mask  = (const int*)  d_in[1];
    const float* W_ih  = (const float*)d_in[2];
    const float* W_hh  = (const float*)d_in[3];
    const float* b_ih  = (const float*)d_in[4];
    const float* b_hh  = (const float*)d_in[5];
    const float* W_lin = (const float*)d_in[6];
    const float* b_lin = (const float*)d_in[7];
    float* out = (float*)d_out;

    tpp_scan_kernel<<<dim3(Bsz / 16), dim3(256), 0, stream>>>(
        times, mask, W_ih, W_hh, b_ih, b_hh, W_lin, b_lin, out);
}
